// Lift2DTo3D_5849745457893
// MI455X (gfx1250) — compile-verified, matrix operand order checked by reading, not run on hardware
//
#include <hip/hip_runtime.h>

// ---- problem constants (from reference setup) ----
#define NZv 10
#define NYv 100
#define NXv 100
#define NVOX (NZv*NYv*NXv)   // 100000
#define BT  2
#define V_  6
#define C_  256
#define H4  112
#define W4  200
#define PV  (H4*W4)          // 22400
#define P_  (V_*PV)          // 134400
#define HH  448
#define WW  800

typedef __attribute__((ext_vector_type(16))) __bf16 v16bf;
typedef __attribute__((ext_vector_type(8)))  float  v8f;

__device__ __forceinline__ bool finitef(float x) {
    // NaN compares false; |inf| > FLT_MAX
    return __builtin_fabsf(x) <= 3.402823466e+38f;
}

// ---------- kernel 1: zero the voxel accumulators (float4 stores) ----------
__global__ void zero_kernel(float4* __restrict__ dst, long n4) {
    long i = (long)blockIdx.x * blockDim.x + threadIdx.x;
    if (i < n4) dst[i] = make_float4(0.f, 0.f, 0.f, 0.f);
}

// ---------- kernel 2: 4x box-downsample + voxelize (per point) ----------
__global__ void prep_kernel(const float* __restrict__ points,
                            const float* __restrict__ conf,
                            float* __restrict__ vol_w,
                            int*   __restrict__ lins,
                            float* __restrict__ wgts) {
    int tid = blockIdx.x * blockDim.x + threadIdx.x;
    if (tid >= BT * P_) return;
    int bt = tid / P_;
    int p  = tid - bt * P_;
    int v  = p / PV;
    int rc = p - v * PV;
    int r  = rc / W4;
    int cc = rc - r * W4;
    int img = bt * V_ + v;
    int R0 = 4*r + 1, C0 = 4*cc + 1;

    float x = 0.f, y = 0.f, z = 0.f, cf = 0.f;
#pragma unroll
    for (int dr = 0; dr < 2; ++dr) {
#pragma unroll
        for (int dc = 0; dc < 2; ++dc) {
            long pix = ((long)img * HH + (R0 + dr)) * WW + (C0 + dc);
            const float* pp = points + pix * 3;
            x += pp[0]; y += pp[1]; z += pp[2];
            cf += conf[pix];
        }
    }
    x *= 0.25f; y *= 0.25f; z *= 0.25f; cf *= 0.25f;

    bool valid = finitef(x) && finitef(y) && finitef(z);
    valid = valid && (cf > 1e-4f);
    valid = valid && (x >= -40.f) && (x < 40.f)
                  && (y >= -40.f) && (y < 40.f)
                  && (z >= -2.f)  && (z < 6.f);

    int lin = 0; float w = 0.f;
    if (valid) {
        int ix = min(max((int)floorf((x + 40.f) * 1.25f), 0), NXv - 1);
        int iy = min(max((int)floorf((y + 40.f) * 1.25f), 0), NYv - 1);
        int iz = min(max((int)floorf((z +  2.f) * 1.25f), 0), NZv - 1);
        lin = iz * (NYv * NXv) + iy * NXv + ix;
        w = cf;
        unsafeAtomicAdd(&vol_w[bt * NVOX + lin], w);   // global_atomic_add_f32
    }
    lins[tid] = lin;
    wgts[tid] = w;
}

// ---------- kernel 3: weighted feature scatter ----------
// idx flattens as [btv, ch, rc] == native feat_1_4 layout -> fully coalesced reads.
__global__ void scatter_kernel(const float* __restrict__ feat,
                               const int*   __restrict__ lins,
                               const float* __restrict__ wgts,
                               float* __restrict__ vol_feat) {
    long idx = (long)blockIdx.x * blockDim.x + threadIdx.x;
    long total = (long)BT * V_ * C_ * PV;
    if (idx >= total) return;
    int rc   = (int)(idx % PV);
    long t1  = idx / PV;
    int ch   = (int)(t1 % C_);
    int btv  = (int)(t1 / C_);
    int bt   = btv / V_;
    int v    = btv - bt * V_;
    int pt   = bt * P_ + v * PV + rc;
    float w  = wgts[pt];
    if (w == 0.f) return;
    int lin = lins[pt];
    float val = feat[idx] * w;
    unsafeAtomicAdd(&vol_feat[((long)(bt * NVOX + lin)) * C_ + ch], val);
}

// ---------- kernel 4: normalize + 1x1 conv as bf16 WMMA GEMM ----------
// One wave per 16-voxel row block; A-fragments (K=256 -> 8 steps) held in
// registers and reused across all 16 N-tiles, so vol_feat streams from HBM once.
__global__ void __launch_bounds__(256)
gemm_kernel(const float* __restrict__ vol_feat,
            const float* __restrict__ vol_w,
            const float* __restrict__ proj_w,
            const float* __restrict__ proj_b,
            float* __restrict__ out) {
    int wave = threadIdx.x >> 5;
    int lane = threadIdx.x & 31;
    int rb = blockIdx.x * 8 + wave;                 // wave-uniform guard (EXEC stays all-1s)
    if (rb >= BT * (NVOX / 16)) return;
    int bt = rb / (NVOX / 16);
    int m0 = (rb - bt * (NVOX / 16)) * 16;
    int mrow = lane & 15;
    int h8   = (lane >> 4) * 8;
    int vidx = m0 + mrow;

    long arow = ((long)(bt * NVOX + vidx)) * C_;
    float invw = 1.0f / fmaxf(vol_w[bt * NVOX + vidx], 1e-6f);

    // A fragments: 16x32 bf16 per K-step, ISA layout:
    // lane (m = lane&15, h = lane>>4); element j -> K = kloc(j) + 8*h (+32*kk)
    v16bf afrag[8];
#pragma unroll
    for (int kk = 0; kk < 8; ++kk) {
#pragma unroll
        for (int j = 0; j < 16; ++j) {
            int vg = j >> 1;
            int kloc = ((vg < 4) ? (vg * 2) : (16 + (vg - 4) * 2)) + (j & 1) + h8;
            afrag[kk][j] = (__bf16)(vol_feat[arow + kk * 32 + kloc] * invw);
        }
    }

    for (int nt = 0; nt < 16; ++nt) {
        int ncol = nt * 16 + mrow;
        long brow = (long)ncol * C_;                // B[k][n] = proj_w[n][k]
        v8f acc = {};
#pragma unroll
        for (int kk = 0; kk < 8; ++kk) {
            v16bf bfrag;
#pragma unroll
            for (int j = 0; j < 16; ++j) {
                int vg = j >> 1;
                int kloc = ((vg < 4) ? (vg * 2) : (16 + (vg - 4) * 2)) + (j & 1) + h8;
                bfrag[j] = (__bf16)proj_w[brow + kk * 32 + kloc];
            }
            acc = __builtin_amdgcn_wmma_f32_16x16x32_bf16(
                false, afrag[kk], false, bfrag, (short)0, acc, false, false);
        }
        float bias = proj_b[ncol];
        // D layout: vgpr r, lane -> m = r + h8, n = lane&15
        long obase = ((long)(bt * C_ + ncol)) * NVOX + m0 + h8;
#pragma unroll
        for (int r = 0; r < 8; ++r) {
            out[obase + r] = acc[r] + bias;
        }
    }
}

extern "C" void kernel_launch(void* const* d_in, const int* in_sizes, int n_in,
                              void* d_out, int out_size, void* d_ws, size_t ws_size,
                              hipStream_t stream) {
    const float* feat   = (const float*)d_in[0];  // [1,2,6,256,112,200]
    const float* points = (const float*)d_in[1];  // [1,2,6,448,800,3]
    const float* conf   = (const float*)d_in[2];  // [1,2,6,448,800]
    const float* proj_w = (const float*)d_in[3];  // [256,256]
    const float* proj_b = (const float*)d_in[4];  // [256]
    float* out = (float*)d_out;                   // [1,2,256,10,100,100]

    // workspace layout (floats): vol_feat | vol_w | lins(int) | wgts
    float* ws = (float*)d_ws;
    long nfeat = (long)BT * NVOX * C_;            // 51,200,000
    long nw    = (long)BT * NVOX;                 //    200,000
    float* vol_feat = ws;
    float* vol_w    = ws + nfeat;
    int*   lins     = (int*)(ws + nfeat + nw);
    float* wgts     = (float*)(lins + BT * P_);

    // 1) zero accumulators
    long n4 = (nfeat + nw) / 4;
    zero_kernel<<<(int)((n4 + 255) / 256), 256, 0, stream>>>((float4*)d_ws, n4);

    // 2) downsample + voxelize
    prep_kernel<<<(BT * P_ + 255) / 256, 256, 0, stream>>>(points, conf, vol_w, lins, wgts);

    // 3) weighted scatter of features
    long total = (long)BT * V_ * C_ * PV;         // 68,812,800
    scatter_kernel<<<(int)((total + 255) / 256), 256, 0, stream>>>(feat, lins, wgts, vol_feat);

    // 4) normalize + projection GEMM (WMMA bf16)
    int nrb = BT * (NVOX / 16);                   // 12,500 waves
    gemm_kernel<<<(nrb + 7) / 8, 256, 0, stream>>>(vol_feat, vol_w, proj_w, proj_b, out);
}
